// CausalSelfAttention_49392123904221
// MI455X (gfx1250) — matrix operand order repaired and checked by reading, not compiled, not run on hardware
//
#include <hip/hip_runtime.h>

// ---------------- CDNA5 WMMA helpers ----------------
typedef __attribute__((ext_vector_type(16))) __bf16 v16bf;
typedef __attribute__((ext_vector_type(8)))  float  v8f;
typedef __attribute__((ext_vector_type(8)))  unsigned short u16x8;
typedef __attribute__((ext_vector_type(4)))  unsigned short u16x4;
typedef __attribute__((ext_vector_type(4)))  int v4i;

#define B_BATCH 4
#define T_SEQ   2048
#define C_DIM   1024
#define H_NUM   16
#define DH      64

#if defined(__has_builtin)
#  if __has_builtin(__builtin_amdgcn_global_load_async_to_lds_b128)
#    define ASYNC_LDS 1
#  endif
#endif

#ifdef ASYNC_LDS
__device__ __forceinline__ void async_copy_b128(const void* g, void* l) {
  // GLOBAL_LOAD_ASYNC_TO_LDS_B128: per-lane 16B global -> LDS, tracked by ASYNCcnt
  __builtin_amdgcn_global_load_async_to_lds_b128(
      (__attribute__((address_space(1))) v4i*)(g),
      (__attribute__((address_space(3))) v4i*)(l), 0, 0);
}
__device__ __forceinline__ void wait_async0() {
  asm volatile("s_wait_asynccnt 0x0" ::: "memory");
}
#else
__device__ __forceinline__ void async_copy_b128(const void* g, void* l) {
  *(u16x8*)l = *(const u16x8*)g;
}
__device__ __forceinline__ void wait_async0() {}
#endif

union FragU { v16bf v; u16x8 h[2]; };

__device__ __forceinline__ v16bf load_frag(const unsigned short* p0,
                                           const unsigned short* p1) {
  FragU f;
  f.h[0] = *(const u16x8*)p0;
  f.h[1] = *(const u16x8*)p1;
  return f.v;
}

__device__ __forceinline__ v8f wmma_bf16(v16bf a, v16bf b, v8f c) {
  return __builtin_amdgcn_wmma_f32_16x16x32_bf16(false, a, false, b, (short)0, c,
                                                 false, false);
}

__device__ __forceinline__ unsigned short f2bf(float x) {
  unsigned u = __builtin_bit_cast(unsigned, x);
  u += 0x7FFFu + ((u >> 16) & 1u);
  return (unsigned short)(u >> 16);
}

// ---------------- Kernel 0a: f32 -> bf16 convert (x) ----------------
__global__ __launch_bounds__(256) void cvt_f32_to_bf16(const float* __restrict__ in,
                                                       unsigned short* __restrict__ out,
                                                       int n4) {
  int i = blockIdx.x * 256 + threadIdx.x;
  if (i >= n4) return;
  float4 f = ((const float4*)in)[i];
  u16x4 h;
  h[0] = f2bf(f.x); h[1] = f2bf(f.y); h[2] = f2bf(f.z); h[3] = f2bf(f.w);
  ((u16x4*)out)[i] = h;
}

// ---------------- Kernel 0b: f32 [K][N] -> bf16 [N][K] transpose ----------------
__global__ __launch_bounds__(256) void transpose_cvt(const float* __restrict__ in,
                                                     unsigned short* __restrict__ out,
                                                     int K, int N) {
  __shared__ unsigned short tile[32][33];
  const int kt = blockIdx.y * 32, nt = blockIdx.x * 32;
  const int tr = threadIdx.x >> 3;         // 0..31
  const int tc = (threadIdx.x & 7) * 4;    // 0,4,...,28
  float4 f = *(const float4*)&in[(size_t)(kt + tr) * N + nt + tc];
  tile[tr][tc + 0] = f2bf(f.x); tile[tr][tc + 1] = f2bf(f.y);
  tile[tr][tc + 2] = f2bf(f.z); tile[tr][tc + 3] = f2bf(f.w);
  __syncthreads();
  u16x4 h;
  h[0] = tile[tc + 0][tr]; h[1] = tile[tc + 1][tr];
  h[2] = tile[tc + 2][tr]; h[3] = tile[tc + 3][tr];
  *(u16x4*)&out[(size_t)(nt + tr) * K + kt + tc] = h;
}

// ---------------- Kernel 1: QKV projection GEMM ----------------
// M=8192, N=3072, K=1024. Block 128 thr, tile 64(M)x128(N), BK=32.
// All staging via async-to-LDS (A from x_bf16, B from pre-transposed W^T).
// Writes Q,K in [B][H][T][DH]; V transposed in [B][H][DH][T] (packed u16x8 rows).
__global__ __launch_bounds__(128) void qkv_gemm(const unsigned short* __restrict__ xb,
                                                const unsigned short* __restrict__ wT,
                                                const float* __restrict__ bias,
                                                unsigned short* __restrict__ qB,
                                                unsigned short* __restrict__ kB,
                                                unsigned short* __restrict__ vT) {
  __shared__ __align__(16) unsigned short As[2][64 * 32];    // [m][k]
  __shared__ __align__(16) unsigned short Bs[2][128 * 32];   // [n][k]
  const int tid  = threadIdx.x;
  const int lane = tid & 31, wave = tid >> 5;
  const int half = lane >> 4, l16 = lane & 15, abase = half * 8;
  const int bn = blockIdx.x * 128, bm = blockIdx.y * 64;
  const int mo = (wave >> 1) * 32, no = (wave & 1) * 64;

  auto stageA = [&](int kt, int bufi) {
#pragma unroll
    for (int it = 0; it < 2; ++it) {
      int i = tid + it * 128;
      int row = i >> 2, c = i & 3;
      async_copy_b128(&xb[(size_t)(bm + row) * C_DIM + kt + c * 8],
                      &As[bufi][row * 32 + c * 8]);
    }
  };
  auto stageB = [&](int kt, int bufi) {
#pragma unroll
    for (int it = 0; it < 4; ++it) {
      int i = tid + it * 128;
      int n = i >> 2, c = i & 3;
      async_copy_b128(&wT[(size_t)(bn + n) * C_DIM + kt + c * 8],
                      &Bs[bufi][n * 32 + c * 8]);
    }
  };

  v8f acc[2][4] = {};

  stageA(0, 0);
  stageB(0, 0);

  const int NSTEP = C_DIM / 32;
  for (int step = 0; step < NSTEP; ++step) {
    const int cur = step & 1, nxt = cur ^ 1;
    const bool hasNext = (step + 1) < NSTEP;
    wait_async0();                               // tile(cur) landed in LDS
    __syncthreads();                             // visible to all waves
    if (hasNext) {                               // async tile(next) overlaps compute
      stageA((step + 1) * 32, nxt);
      stageB((step + 1) * 32, nxt);
    }

    v16bf af[2], bf4[4];
#pragma unroll
    for (int mf = 0; mf < 2; ++mf) {
      int row = mo + mf * 16 + l16;
      af[mf] = load_frag(&As[cur][row * 32 + abase],
                         &As[cur][row * 32 + 16 + abase]);
    }
#pragma unroll
    for (int nf = 0; nf < 4; ++nf) {
      int n = no + nf * 16 + l16;
      bf4[nf] = load_frag(&Bs[cur][n * 32 + abase],
                          &Bs[cur][n * 32 + 16 + abase]);
    }
#pragma unroll
    for (int mf = 0; mf < 2; ++mf)
#pragma unroll
      for (int nf = 0; nf < 4; ++nf)
        acc[mf][nf] = wmma_bf16(af[mf], bf4[nf], acc[mf][nf]);
  }

  // epilogue: +bias; Q,K scatter bf16; V packed u16x8 into [B][H][DH][T]
#pragma unroll
  for (int mf = 0; mf < 2; ++mf)
#pragma unroll
    for (int nf = 0; nf < 4; ++nf) {
      int n = bn + no + nf * 16 + l16;
      float bia = bias[n];
      int which = n >> 10;
      int h = (n >> 6) & 15;
      int d = n & 63;
      int m0 = bm + mo + mf * 16 + half * 8;     // 8 consecutive rows
      int b = m0 >> 11, t0 = m0 & (T_SEQ - 1);
      if (which == 2) {
        u16x8 pk;
#pragma unroll
        for (int r = 0; r < 8; ++r) pk[r] = f2bf(acc[mf][nf][r] + bia);
        *(u16x8*)&vT[((size_t)(b * H_NUM + h) * DH + d) * T_SEQ + t0] = pk;
      } else {
        unsigned short* dst = (which == 0) ? qB : kB;
#pragma unroll
        for (int r = 0; r < 8; ++r)
          dst[((size_t)(b * H_NUM + h) * T_SEQ + t0 + r) * DH + d] =
              f2bf(acc[mf][nf][r] + bia);
      }
    }
}

// ---------------- Kernel 2: causal flash attention ----------------
// Block 128 thr = 4 waves, 64 q-rows/block, per (b,h). Async double-buffered K/V^T.
__global__ __launch_bounds__(128) void flash_attn(const unsigned short* __restrict__ qB,
                                                  const unsigned short* __restrict__ kB,
                                                  const unsigned short* __restrict__ vT,
                                                  unsigned short* __restrict__ o) {
  __shared__ __align__(16) unsigned short Ks[2][32 * 64];   // [key][d]
  __shared__ __align__(16) unsigned short Vt[2][64 * 32];   // [d][key]
  __shared__ __align__(16) unsigned short Ps[4][16 * 32];   // per-wave P tile
  const int tid  = threadIdx.x;
  const int lane = tid & 31, wave = tid >> 5;
  const int half = lane >> 4, l16 = lane & 15, abase = half * 8;
  const int qb0 = blockIdx.x * 64;
  const int bh  = blockIdx.y;
  const int b = bh >> 4, h = bh & 15;
  const unsigned short* Qp = qB + (size_t)bh * T_SEQ * DH;
  const unsigned short* Kp = kB + (size_t)bh * T_SEQ * DH;
  const unsigned short* Vp = vT + (size_t)bh * DH * T_SEQ;
  const int q0 = qb0 + wave * 16;
  const int qrow = q0 + l16;

  auto stageK = [&](int kt, int bufi) {
#pragma unroll
    for (int it = 0; it < 2; ++it) {
      int i = tid + it * 128;
      int r = i >> 3, c = i & 7;
      async_copy_b128(&Kp[(size_t)(kt + r) * DH + c * 8],
                      &Ks[bufi][r * 64 + c * 8]);
    }
  };
  auto stageVt = [&](int kt, int bufi) {
#pragma unroll
    for (int it = 0; it < 2; ++it) {
      int i = tid + it * 128;
      int d = i >> 2, c = i & 3;
      async_copy_b128(&Vp[(size_t)d * T_SEQ + kt + c * 8],
                      &Vt[bufi][d * 32 + c * 8]);
    }
  };

  v16bf qf[2];
  qf[0] = load_frag(&Qp[qrow * DH + abase],      &Qp[qrow * DH + 16 + abase]);
  qf[1] = load_frag(&Qp[qrow * DH + 32 + abase], &Qp[qrow * DH + 48 + abase]);

  v8f Ob[4] = {};
  float mrow[8], lrow[8];
#pragma unroll
  for (int r = 0; r < 8; ++r) { mrow[r] = -3.0e38f; lrow[r] = 0.0f; }

  const float sl2 = 0.125f * 1.44269504088896340736f; // (1/sqrt(Dh))*log2(e)

  stageK(0, 0);
  stageVt(0, 0);

  const int nsteps = (qb0 + 64) / 32;   // uniform across block
  for (int step = 0; step < nsteps; ++step) {
    const int kt = step * 32;
    const int cur = step & 1, nxt = cur ^ 1;
    const bool hasNext = (step + 1) < nsteps;
    wait_async0();
    __syncthreads();
    if (hasNext) {
      stageK(kt + 32, nxt);
      stageVt(kt + 32, nxt);
    }

    // S = Q . K^T
    v8f s[2];
#pragma unroll
    for (int nb = 0; nb < 2; ++nb) {
      v8f acc = {};
#pragma unroll
      for (int db = 0; db < 2; ++db) {
        const unsigned short* base =
            &Ks[cur][(nb * 16 + l16) * 64 + db * 32 + abase];
        acc = wmma_bf16(qf[db], load_frag(base, base + 16), acc);
      }
      s[nb] = acc;
    }

    // causal mask + log2-domain scale
#pragma unroll
    for (int nb = 0; nb < 2; ++nb) {
      int key = kt + nb * 16 + l16;
#pragma unroll
      for (int r = 0; r < 8; ++r) {
        int q = q0 + r + half * 8;
        float v = s[nb][r] * sl2;
        s[nb][r] = (key <= q) ? v : -3.0e38f;
      }
    }

    // online softmax (rows span 16-lane halves)
#pragma unroll
    for (int r = 0; r < 8; ++r) {
      float mx = fmaxf(s[0][r], s[1][r]);
#pragma unroll
      for (int off = 8; off >= 1; off >>= 1)
        mx = fmaxf(mx, __shfl_xor(mx, off, 16));
      float mnew = fmaxf(mrow[r], mx);
      float corr = exp2f(mrow[r] - mnew);
      mrow[r] = mnew;
      float p0 = exp2f(s[0][r] - mnew);
      float p1 = exp2f(s[1][r] - mnew);
      s[0][r] = p0; s[1][r] = p1;
      float rs = p0 + p1;
#pragma unroll
      for (int off = 8; off >= 1; off >>= 1)
        rs += __shfl_xor(rs, off, 16);
      lrow[r] = lrow[r] * corr + rs;
#pragma unroll
      for (int db = 0; db < 4; ++db) Ob[db][r] *= corr;
    }

    // re-layout P (C/D layout) -> A-fragment via per-wave LDS tile
    unsigned short* P = &Ps[wave][0];
#pragma unroll
    for (int nb = 0; nb < 2; ++nb)
#pragma unroll
      for (int r = 0; r < 8; ++r)
        P[(r + half * 8) * 32 + nb * 16 + l16] = f2bf(s[nb][r]);
    asm volatile("s_wait_dscnt 0x0" ::: "memory");
    v16bf pf = load_frag(&P[l16 * 32 + abase], &P[l16 * 32 + 16 + abase]);

    // O += P . V
#pragma unroll
    for (int db = 0; db < 4; ++db) {
      const unsigned short* base = &Vt[cur][(db * 16 + l16) * 32 + abase];
      Ob[db] = wmma_bf16(pf, load_frag(base, base + 16), Ob[db]);
    }
  }

  // normalize, store bf16 [B][T][H*DH]
#pragma unroll
  for (int r = 0; r < 8; ++r) {
    float inv = 1.0f / lrow[r];
    int q = q0 + r + half * 8;
    unsigned short* dst = o + ((size_t)b * T_SEQ + q) * C_DIM + h * DH;
#pragma unroll
    for (int db = 0; db < 4; ++db)
      dst[db * 16 + l16] = f2bf(Ob[db][r] * inv);
  }
}

// ---------------- Kernel 3: output projection GEMM ----------------
// M=8192, N=1024, K=1024. A = attn out (bf16), B = pre-transposed W_out^T.
__global__ __launch_bounds__(128) void out_gemm(const unsigned short* __restrict__ ab,
                                                const unsigned short* __restrict__ wT,
                                                const float* __restrict__ bias,
                                                float* __restrict__ out) {
  __shared__ __align__(16) unsigned short As[2][64 * 32];
  __shared__ __align__(16) unsigned short Bs[2][128 * 32];
  const int tid  = threadIdx.x;
  const int lane = tid & 31, wave = tid >> 5;
  const int half = lane >> 4, l16 = lane & 15, abase = half * 8;
  const int bn = blockIdx.x * 128, bm = blockIdx.y * 64;
  const int mo = (wave >> 1) * 32, no = (wave & 1) * 64;

  auto stageA = [&](int kt, int bufi) {
#pragma unroll
    for (int it = 0; it < 2; ++it) {
      int i = tid + it * 128;
      int row = i >> 2, c = i & 3;
      async_copy_b128(&ab[(size_t)(bm + row) * C_DIM + kt + c * 8],
                      &As[bufi][row * 32 + c * 8]);
    }
  };
  auto stageB = [&](int kt, int bufi) {
#pragma unroll
    for (int it = 0; it < 4; ++it) {
      int i = tid + it * 128;
      int n = i >> 2, c = i & 3;
      async_copy_b128(&wT[(size_t)(bn + n) * C_DIM + kt + c * 8],
                      &Bs[bufi][n * 32 + c * 8]);
    }
  };

  v8f acc[2][4] = {};

  stageA(0, 0);
  stageB(0, 0);

  const int NSTEP = C_DIM / 32;
  for (int step = 0; step < NSTEP; ++step) {
    const int cur = step & 1, nxt = cur ^ 1;
    const bool hasNext = (step + 1) < NSTEP;
    wait_async0();
    __syncthreads();
    if (hasNext) {
      stageA((step + 1) * 32, nxt);
      stageB((step + 1) * 32, nxt);
    }

    v16bf af[2], bf4[4];
#pragma unroll
    for (int mf = 0; mf < 2; ++mf) {
      int row = mo + mf * 16 + l16;
      af[mf] = load_frag(&As[cur][row * 32 + abase],
                         &As[cur][row * 32 + 16 + abase]);
    }
#pragma unroll
    for (int nf = 0; nf < 4; ++nf) {
      int n = no + nf * 16 + l16;
      bf4[nf] = load_frag(&Bs[cur][n * 32 + abase],
                          &Bs[cur][n * 32 + 16 + abase]);
    }
#pragma unroll
    for (int mf = 0; mf < 2; ++mf)
#pragma unroll
      for (int nf = 0; nf < 4; ++nf)
        acc[mf][nf] = wmma_bf16(af[mf], bf4[nf], acc[mf][nf]);
  }

#pragma unroll
  for (int mf = 0; mf < 2; ++mf)
#pragma unroll
    for (int nf = 0; nf < 4; ++nf) {
      int n = bn + no + nf * 16 + l16;
      float bia = bias[n];
#pragma unroll
      for (int r = 0; r < 8; ++r) {
        int m = bm + mo + mf * 16 + r + half * 8;
        out[(size_t)m * C_DIM + n] = acc[mf][nf][r] + bia;
      }
    }
}

// ---------------- launcher ----------------
extern "C" void kernel_launch(void* const* d_in, const int* in_sizes, int n_in,
                              void* d_out, int out_size, void* d_ws, size_t ws_size,
                              hipStream_t stream) {
  const float* x     = (const float*)d_in[0];
  const float* w_qkv = (const float*)d_in[1];
  const float* b_qkv = (const float*)d_in[2];
  const float* w_out = (const float*)d_in[3];
  const float* b_out = (const float*)d_in[4];
  float* out = (float*)d_out;

  char* ws = (char*)d_ws;
  const size_t MB = 1024 * 1024;
  unsigned short* xb    = (unsigned short*)(ws + 0);          // 16 MB  x bf16
  unsigned short* wqkvT = (unsigned short*)(ws + 16 * MB);    //  6 MB  W_qkv^T bf16 [3072][1024]
  unsigned short* woutT = (unsigned short*)(ws + 22 * MB);    //  2 MB  W_out^T bf16 [1024][1024]
  unsigned short* qB    = (unsigned short*)(ws + 24 * MB);    // 16 MB  [B][H][T][DH]
  unsigned short* kB    = (unsigned short*)(ws + 40 * MB);    // 16 MB  [B][H][T][DH]
  unsigned short* vT    = (unsigned short*)(ws + 56 * MB);    // 16 MB  [B][H][DH][T]
  unsigned short* attnO = (unsigned short*)(ws + 72 * MB);    // 16 MB  [B][T][C] bf16

  int nx = B_BATCH * T_SEQ * C_DIM;
  cvt_f32_to_bf16<<<(nx / 4 + 255) / 256, 256, 0, stream>>>(x, xb, nx / 4);
  transpose_cvt<<<dim3(3 * C_DIM / 32, C_DIM / 32), 256, 0, stream>>>(
      w_qkv, wqkvT, C_DIM, 3 * C_DIM);
  transpose_cvt<<<dim3(C_DIM / 32, C_DIM / 32), 256, 0, stream>>>(
      w_out, woutT, C_DIM, C_DIM);

  qkv_gemm<<<dim3(3 * C_DIM / 128, B_BATCH * T_SEQ / 64), 128, 0, stream>>>(
      xb, wqkvT, b_qkv, qB, kB, vT);

  flash_attn<<<dim3(T_SEQ / 64, B_BATCH * H_NUM), 128, 0, stream>>>(qB, kB, vT, attnO);

  out_gemm<<<dim3(C_DIM / 128, B_BATCH * T_SEQ / 64), 128, 0, stream>>>(
      attnO, woutT, b_out, out);
}
